// DeepseekBlock_16501264351756
// MI455X (gfx1250) — matrix operand order repaired and checked by reading, not compiled, not run on hardware
//
#include <hip/hip_runtime.h>
#include <hip/hip_bf16.h>

// ---------------- problem constants ----------------
constexpr int B_  = 4;
constexpr int S_  = 1024;
constexpr int H_  = 1024;
constexpr int NH_ = 16;
constexpr int L_  = 256;
constexpr int RD_ = 64;
constexpr int DF_ = 2048;
constexpr int E_  = 8;
constexpr int NT_ = B_ * S_;          // 4096 tokens
constexpr float LN_EPS_ = 1e-5f;
constexpr float SCALE_  = 0.125f;     // HD^-0.5

typedef __attribute__((ext_vector_type(16))) __bf16 v16bf;
typedef __attribute__((ext_vector_type(8)))  float  v8f;
typedef __attribute__((ext_vector_type(4)))  unsigned int v4u;
typedef __attribute__((ext_vector_type(8)))  int v8i;
typedef __attribute__((ext_vector_type(4)))  int v4i;

union FragU {
    v16bf v;
    unsigned int   u[8];
    unsigned short s[16];
};

__device__ __forceinline__ unsigned short f2bf(float x) {
    unsigned int u = __float_as_uint(x);
    unsigned int r = u + 0x7FFFu + ((u >> 16) & 1u);   // RNE
    return (unsigned short)(r >> 16);
}

// =====================================================================
// Tensor Data Mover: issue a 2D fp32 tile load (global -> LDS).
//   lds_off    : byte offset into LDS (dynamic LDS assumed to start at 0)
//   gaddr      : global address of the TILE start
//   tensor_d0/1: remaining tensor extent from tile start (elements) - the
//                TDM zero-fills reads past these (handles M/N edges + the
//                device-side-count MoE GEMMs)
//   tile_d0/1  : tile shape (d0 = contiguous dim)
//   stride     : elements between consecutive dim1 lines
// =====================================================================
#if __has_builtin(__builtin_amdgcn_tensor_load_to_lds)
#define HAVE_TDM 1
__device__ __forceinline__ void tdm_load_2d_f32(
    unsigned lds_off, const float* gaddr,
    unsigned tensor_d0, unsigned tensor_d1,
    unsigned tile_d0, unsigned tile_d1, unsigned stride)
{
    unsigned long long ga = (unsigned long long)(uintptr_t)gaddr;
    v4u g0;
    g0.x = 1u;                                       // count=1 (valid), no gather
    g0.y = lds_off;                                  // lds_addr [63:32]
    g0.z = (unsigned)(ga & 0xFFFFFFFFu);             // global_addr [95:64]
    g0.w = (unsigned)((ga >> 32) & 0x01FFFFFFu)      // global_addr [120:96]
         | 0x80000000u;                              // type=2 ("image") [127:126]
    v8i g1;
    g1[0] = (int)(2u << 16);                         // data_size=2 (4B); mask=0
    g1[1] = (int)((tensor_d0 & 0xFFFFu) << 16);      // tensor_dim0[15:0] @ [63:48]
    g1[2] = (int)((tensor_d0 >> 16) | ((tensor_d1 & 0xFFFFu) << 16)); // d0 hi, d1 lo
    g1[3] = (int)((tensor_d1 >> 16) | (tile_d0 << 16));               // d1 hi, tile_dim0
    g1[4] = (int)(tile_d1 & 0xFFFFu);                // tile_dim1 ; tile_dim2=0
    g1[5] = (int)stride;                             // tensor_dim0_stride[31:0]
    g1[6] = 0;                                       // stride hi / dim1_stride lo
    g1[7] = 0;
    v4i z4 = {0, 0, 0, 0};
    v8i z8 = {0, 0, 0, 0, 0, 0, 0, 0};
    // clang-23 / therock-10.0 arity: (g0, g1, g2, g3, g4, cpol)
    __builtin_amdgcn_tensor_load_to_lds(g0, g1, z4, z4, z8, 0);
}
__device__ __forceinline__ void tdm_wait() {
#if __has_builtin(__builtin_amdgcn_s_wait_tensorcnt)
    __builtin_amdgcn_s_wait_tensorcnt(0);
#else
    asm volatile("s_wait_tensorcnt 0x0" ::: "memory");
#endif
}
#endif

// =====================================================================
// WMMA bf16 GEMM:  C[M,N] = A[M,K] @ B[K,N]  (fp32 in memory, TDM-staged
// fp32 tiles in LDS, bf16 conversion at fragment build, fp32 accumulate).
// Block = 128 thr = 4 waves (2x2); tile 64x128; wave tile 32x64 = 8 WMMA.
// m_dev: optional device-side row count (gathered MoE GEMMs).
// =====================================================================
#define GBM 64
#define GBN 128
#define GBK 32
#define SB_OFF (GBM * GBK)          // B tile starts after A tile (floats)

__global__ __launch_bounds__(128) void gemm_bf16_wmma(
    const float* __restrict__ A, const float* __restrict__ B,
    float* __restrict__ C, int M, int N, int K,
    int lda, int ldb, int ldc, const int* __restrict__ m_dev)
{
    if (m_dev) { int md = m_dev[0]; M = M < md ? M : md; }
    int tile_m = blockIdx.y * GBM;
    int tile_n = blockIdx.x * GBN;
    if (tile_m >= M) return;   // uniform per block

    extern __shared__ float smem[];          // [0,2048): A 64x32 ; [2048,6144): B 32x128
    const float* sA = smem;
    const float* sB = smem + SB_OFF;

    int tid  = threadIdx.x;
    int wid  = tid >> 5;
    int lane = tid & 31;
    int l15   = lane & 15;
    int lhalf = lane >> 4;
    int kb8  = lhalf * 8;
    int kb16 = lhalf * 16;
    int wrow = (wid >> 1) * 32;
    int wcol = (wid & 1) * 64;

    v8f acc[2][4] = {};

    for (int k0 = 0; k0 < K; k0 += GBK) {
#ifdef HAVE_TDM
        if (tid < 32) {   // one wave issues both tile DMAs
            tdm_load_2d_f32(0, A + (size_t)tile_m * lda + k0,
                            (unsigned)(K - k0), (unsigned)(M - tile_m),
                            GBK, GBM, (unsigned)lda);
            tdm_load_2d_f32(SB_OFF * 4, B + (size_t)k0 * ldb + tile_n,
                            (unsigned)(N - tile_n), (unsigned)(K - k0),
                            GBN, GBK, (unsigned)ldb);
            tdm_wait();
        }
        __syncthreads();
#else
        for (int i = tid; i < GBM * GBK; i += 128) {
            int r = i >> 5, c = i & 31;
            int gr = tile_m + r;
            smem[i] = (gr < M) ? A[(size_t)gr * lda + k0 + c] : 0.f;
        }
        for (int i = tid; i < GBK * GBN; i += 128) {
            int r = i >> 7, c = i & 127;
            int gn = tile_n + c;
            smem[SB_OFF + i] = (gn < N) ? B[(size_t)(k0 + r) * ldb + gn] : 0.f;
        }
        __syncthreads();
#endif
        // ---- fragments (fp32 LDS -> bf16 regs) ----
        FragU fa[2], fb[4];
        #pragma unroll
        for (int i = 0; i < 2; ++i) {
            const float* rp = sA + (size_t)(wrow + i * 16 + l15) * GBK;
            #pragma unroll
            for (int e = 0; e < 8; ++e) {
                int K2 = kb16 + 2 * e;
                fa[i].s[2 * e]     = f2bf(rp[K2]);
                fa[i].s[2 * e + 1] = f2bf(rp[K2 + 1]);
            }
        }
        #pragma unroll
        for (int j = 0; j < 4; ++j) {
            int col = wcol + j * 16 + l15;
            #pragma unroll
            for (int e = 0; e < 16; ++e)
                fb[j].s[e] = f2bf(sB[(size_t)(kb16 + e) * GBN + col]);
        }
        #pragma unroll
        for (int i = 0; i < 2; ++i)
            #pragma unroll
            for (int j = 0; j < 4; ++j)
                acc[i][j] = __builtin_amdgcn_wmma_f32_16x16x32_bf16(
                    false, fa[i].v, false, fb[j].v, (short)0, acc[i][j], false, false);
        __syncthreads();
    }

    // ---- store C ----
    #pragma unroll
    for (int i = 0; i < 2; ++i)
        #pragma unroll
        for (int j = 0; j < 4; ++j)
            #pragma unroll
            for (int r = 0; r < 8; ++r) {
                int row = tile_m + wrow + i * 16 + r + lhalf * 8;
                int col = tile_n + wcol + j * 16 + l15;
                if (row < M && col < N)
                    C[(size_t)row * ldc + col] = acc[i][j][r];
            }
}

// =====================================================================
// LayerNorm: one block (256 thr) per token.
// =====================================================================
__global__ __launch_bounds__(256) void ln_kernel(
    const float* __restrict__ x, const float* __restrict__ g,
    const float* __restrict__ b, float* __restrict__ y)
{
    int t = blockIdx.x;
    const float* xr = x + (size_t)t * H_;
    float* yr = y + (size_t)t * H_;
    __shared__ float sh[8];
    int lane = threadIdx.x & 31, wid = threadIdx.x >> 5;

    float s = 0.f;
    for (int i = threadIdx.x; i < H_; i += 256) s += xr[i];
    #pragma unroll
    for (int o = 16; o; o >>= 1) s += __shfl_xor(s, o, 32);
    if (lane == 0) sh[wid] = s;
    __syncthreads();
    float mean = 0.f;
    #pragma unroll
    for (int i = 0; i < 8; ++i) mean += sh[i];
    mean *= (1.f / H_);
    __syncthreads();

    float v = 0.f;
    for (int i = threadIdx.x; i < H_; i += 256) { float d = xr[i] - mean; v += d * d; }
    #pragma unroll
    for (int o = 16; o; o >>= 1) v += __shfl_xor(v, o, 32);
    if (lane == 0) sh[wid] = v;
    __syncthreads();
    float var = 0.f;
    #pragma unroll
    for (int i = 0; i < 8; ++i) var += sh[i];
    var *= (1.f / H_);
    float inv = rsqrtf(var + LN_EPS_);
    for (int i = threadIdx.x; i < H_; i += 256)
        yr[i] = (xr[i] - mean) * inv * g[i] + b[i];
}

// =====================================================================
// Pack q = [q_c | rope(q_r)] * SCALE into (B*NH, S, 128)
// =====================================================================
__global__ __launch_bounds__(256) void pack_q_kernel(
    const float* __restrict__ qc, const float* __restrict__ qr,
    float* __restrict__ q)
{
    int t = blockIdx.x;                // token
    int b = t / S_, s = t % S_;
    for (int idx = threadIdx.x; idx < NH_ * 64; idx += 256) {
        int h = idx >> 6, j = idx & 63;
        float* qd = q + (((size_t)(b * NH_ + h) * S_) + s) * 128;
        if (j < 32) {
            qd[2 * j]     = qc[(size_t)t * H_ + h * 64 + 2 * j] * SCALE_;
            qd[2 * j + 1] = qc[(size_t)t * H_ + h * 64 + 2 * j + 1] * SCALE_;
        } else {
            int i = j - 32;
            float xe = qr[(size_t)t * H_ + h * 64 + 2 * i];
            float xo = qr[(size_t)t * H_ + h * 64 + 2 * i + 1];
            float invf = __powf(10000.f, -(2.f * i) / (float)RD_);
            float sn, cs; __sincosf((float)s * invf, &sn, &cs);
            qd[64 + 2 * i]     = (xe * cs - xo * sn) * SCALE_;
            qd[64 + 2 * i + 1] = (xe * sn + xo * cs) * SCALE_;
        }
    }
}

// =====================================================================
// Pack k = [k_c | rope(k_r)] into (B*NH, S, 128) and v into (B*NH, S, 64)
// =====================================================================
__global__ __launch_bounds__(256) void pack_kv_kernel(
    const float* __restrict__ kv, const float* __restrict__ kr,
    float* __restrict__ k, float* __restrict__ v)
{
    int t = blockIdx.x;
    int b = t / S_, s = t % S_;
    for (int idx = threadIdx.x; idx < NH_ * 64; idx += 256) {
        int h = idx >> 6, j = idx & 63;
        float* kd = k + (((size_t)(b * NH_ + h) * S_) + s) * 128;
        float* vd = v + (((size_t)(b * NH_ + h) * S_) + s) * 64;
        if (j < 32) {
            kd[2 * j]     = kv[(size_t)t * (2 * H_) + h * 64 + 2 * j];
            kd[2 * j + 1] = kv[(size_t)t * (2 * H_) + h * 64 + 2 * j + 1];
            vd[2 * j]     = kv[(size_t)t * (2 * H_) + H_ + h * 64 + 2 * j];
            vd[2 * j + 1] = kv[(size_t)t * (2 * H_) + H_ + h * 64 + 2 * j + 1];
        } else {
            int i = j - 32;
            float xe = kr[(size_t)t * RD_ + 2 * i];
            float xo = kr[(size_t)t * RD_ + 2 * i + 1];
            float invf = __powf(10000.f, -(2.f * i) / (float)RD_);
            float sn, cs; __sincosf((float)s * invf, &sn, &cs);
            kd[64 + 2 * i]     = xe * cs - xo * sn;
            kd[64 + 2 * i + 1] = xe * sn + xo * cs;
        }
    }
}

// =====================================================================
// Flash attention: 1 wave = 16 query rows; block = 4 waves.
// grid = (B*NH, S/64). WMMA bf16 for Q.K^T and P.V; online softmax.
// =====================================================================
__global__ __launch_bounds__(128) void attn_kernel(
    const float* __restrict__ q, const float* __restrict__ k,
    const float* __restrict__ v, float* __restrict__ o)
{
    int bh = blockIdx.x;
    int b = bh / NH_, h = bh % NH_;
    int wid  = threadIdx.x >> 5;
    int lane = threadIdx.x & 31;
    int l15   = lane & 15;
    int lhalf = lane >> 4;
    int qbase = blockIdx.y * 64 + wid * 16;
    int kb8 = lhalf * 8;

    __shared__ unsigned short sP[4][16][32];

    const float* qb = q + (size_t)bh * S_ * 128;
    const float* kb_ = k + (size_t)bh * S_ * 128;
    const float* vb = v + (size_t)bh * S_ * 64;

    // Q fragments (4 x 16x32), kept in registers, pre-scaled
    FragU qa[4];
    #pragma unroll
    for (int c = 0; c < 4; ++c) {
        const float* p = qb + ((size_t)(qbase + l15)) * 128 + c * 32;
        #pragma unroll
        for (int e = 0; e < 8; ++e) {
            int K2 = lhalf * 16 + 2 * e;
            qa[c].s[2 * e]     = f2bf(p[K2]);
            qa[c].s[2 * e + 1] = f2bf(p[K2 + 1]);
        }
    }

    float m[8], l[8];
    v8f accv[4] = {};
    #pragma unroll
    for (int r = 0; r < 8; ++r) { m[r] = -INFINITY; l[r] = 0.f; }

    int kend = qbase + 16;
    for (int kb = 0; kb < kend; kb += 32) {
        v8f s0 = {}, s1 = {};
        int key0 = kb + l15;
        int key1 = kb + 16 + l15;
        int ck0 = key0 < (S_ - 1) ? key0 : (S_ - 1);
        int ck1 = key1 < (S_ - 1) ? key1 : (S_ - 1);
        #pragma unroll
        for (int c = 0; c < 4; ++c) {
            FragU b0, b1;
            const float* kp0 = kb_ + (size_t)ck0 * 128 + c * 32 + lhalf * 16;
            const float* kp1 = kb_ + (size_t)ck1 * 128 + c * 32 + lhalf * 16;
            #pragma unroll
            for (int e = 0; e < 16; ++e) { b0.s[e] = f2bf(kp0[e]); b1.s[e] = f2bf(kp1[e]); }
            s0 = __builtin_amdgcn_wmma_f32_16x16x32_bf16(false, qa[c].v, false, b0.v, (short)0, s0, false, false);
            s1 = __builtin_amdgcn_wmma_f32_16x16x32_bf16(false, qa[c].v, false, b1.v, (short)0, s1, false, false);
        }

        // online softmax per row; write P tile to LDS (bf16)
        #pragma unroll
        for (int r = 0; r < 8; ++r) {
            int row = qbase + r + lhalf * 8;
            float v0 = (key0 <= row) ? s0[r] : -INFINITY;
            float v1 = (key1 <= row) ? s1[r] : -INFINITY;
            float mx = fmaxf(v0, v1);
            #pragma unroll
            for (int off = 1; off < 16; off <<= 1) mx = fmaxf(mx, __shfl_xor(mx, off, 32));
            float mn = fmaxf(m[r], mx);
            float corr = __expf(m[r] - mn);
            m[r] = mn;
            float p0 = __expf(v0 - mn);
            float p1 = __expf(v1 - mn);
            float sum = p0 + p1;
            #pragma unroll
            for (int off = 1; off < 16; off <<= 1) sum += __shfl_xor(sum, off, 32);
            l[r] = l[r] * corr + sum;
            #pragma unroll
            for (int g = 0; g < 4; ++g) accv[g][r] = accv[g][r] * corr;
            int prow = r + lhalf * 8;
            sP[wid][prow][l15]      = f2bf(p0);
            sP[wid][prow][16 + l15] = f2bf(p1);
        }

        // P as A fragment (same-wave LDS RAW: DS ops in-order per wave)
        FragU pa;
        {
            const unsigned short* rp = &sP[wid][l15][0];
            #pragma unroll
            for (int e = 0; e < 8; ++e) {
                int K2 = lhalf * 16 + 2 * e;
                pa.s[2 * e]     = rp[K2];
                pa.s[2 * e + 1] = rp[K2 + 1];
            }
        }
        // V fragments + accumulate
        #pragma unroll
        for (int g = 0; g < 4; ++g) {
            FragU vbF;
            int dim = g * 16 + l15;
            #pragma unroll
            for (int e = 0; e < 16; ++e) {
                int key = kb + lhalf * 16 + e;
                key = key < (S_ - 1) ? key : (S_ - 1);
                vbF.s[e] = f2bf(vb[(size_t)key * 64 + dim]);
            }
            accv[g] = __builtin_amdgcn_wmma_f32_16x16x32_bf16(false, pa.v, false, vbF.v, (short)0, accv[g], false, false);
        }
    }

    // store o in (B, S, H) layout for the W_o GEMM
    #pragma unroll
    for (int r = 0; r < 8; ++r) {
        int row = qbase + r + lhalf * 8;
        float inv = 1.f / l[r];
        #pragma unroll
        for (int g = 0; g < 4; ++g) {
            int dim = g * 16 + l15;
            o[((size_t)b * S_ + row) * H_ + h * 64 + dim] = accv[g][r] * inv;
        }
    }
}

// =====================================================================
// Residual add: out = x + y
// =====================================================================
__global__ void add_kernel(const float* __restrict__ x, const float* __restrict__ y,
                           float* __restrict__ out, size_t n)
{
    for (size_t i = (size_t)blockIdx.x * blockDim.x + threadIdx.x; i < n;
         i += (size_t)gridDim.x * blockDim.x)
        out[i] = x[i] + y[i];
}

// =====================================================================
// Routing: block (256 thr = 8 waves) per token; wave e computes sigmoid
// affinity to centroid e; thread 0 does top-2 + appends to expert lists.
// =====================================================================
__global__ __launch_bounds__(256) void route_kernel(
    const float* __restrict__ xf, const float* __restrict__ cent,
    const float* __restrict__ rbias, int* __restrict__ ecnt,
    int* __restrict__ elist, float* __restrict__ ewgt)
{
    int t = blockIdx.x;
    int wid = threadIdx.x >> 5, lane = threadIdx.x & 31;
    __shared__ float sc[E_];
    const float* xr = xf + (size_t)t * H_;
    const float* cr = cent + (size_t)wid * H_;
    float s = 0.f;
    for (int i = lane; i < H_; i += 32) s += xr[i] * cr[i];
    #pragma unroll
    for (int o = 16; o; o >>= 1) s += __shfl_xor(s, o, 32);
    if (lane == 0) sc[wid] = 1.f / (1.f + __expf(-s)) + rbias[wid];
    __syncthreads();
    if (threadIdx.x == 0) {
        int i0 = 0; float b0 = sc[0];
        #pragma unroll
        for (int e = 1; e < E_; ++e) if (sc[e] > b0) { b0 = sc[e]; i0 = e; }
        int i1 = (i0 == 0) ? 1 : 0; float b1 = sc[i1];
        #pragma unroll
        for (int e = 0; e < E_; ++e)
            if (e != i0 && sc[e] > b1) { b1 = sc[e]; i1 = e; }
        float mx = fmaxf(b0, b1);
        float e0 = __expf(b0 - mx), e1 = __expf(b1 - mx);
        float inv = 1.f / (e0 + e1);
        int p0 = atomicAdd(&ecnt[i0], 1);
        elist[i0 * NT_ + p0] = t; ewgt[i0 * NT_ + p0] = e0 * inv;
        int p1 = atomicAdd(&ecnt[i1], 1);
        elist[i1 * NT_ + p1] = t; ewgt[i1 * NT_ + p1] = e1 * inv;
    }
}

// Gather rows for expert e
__global__ __launch_bounds__(256) void gather_kernel(
    const float* __restrict__ xf, const int* __restrict__ elist,
    const int* __restrict__ ecnt, int e, float* __restrict__ Xg)
{
    int r = blockIdx.x;
    if (r >= ecnt[e]) return;
    int t = elist[e * NT_ + r];
    for (int i = threadIdx.x; i < H_; i += 256)
        Xg[(size_t)r * H_ + i] = xf[(size_t)t * H_ + i];
}

// G = silu(G) * U (only first cnt rows matter)
__global__ void silu_mul_kernel(float* __restrict__ G, const float* __restrict__ U,
                                const int* __restrict__ ecnt, int e)
{
    size_t n = (size_t)ecnt[e] * DF_;
    for (size_t i = (size_t)blockIdx.x * blockDim.x + threadIdx.x; i < n;
         i += (size_t)gridDim.x * blockDim.x) {
        float g = G[i];
        G[i] = (g / (1.f + __expf(-g))) * U[i];
    }
}

// Scatter-add weighted expert output into residual stream
__global__ __launch_bounds__(256) void scatter_kernel(
    const float* __restrict__ OutG, const int* __restrict__ elist,
    const float* __restrict__ ewgt, const int* __restrict__ ecnt,
    int e, float* __restrict__ out)
{
    int r = blockIdx.x;
    if (r >= ecnt[e]) return;
    int t = elist[e * NT_ + r];
    float w = ewgt[e * NT_ + r];
    for (int i = threadIdx.x; i < H_; i += 256)
        out[(size_t)t * H_ + i] += w * OutG[(size_t)r * H_ + i];
}

__global__ void counts_out_kernel(const int* __restrict__ ecnt, float* __restrict__ out)
{
    if (threadIdx.x < E_) out[threadIdx.x] = (float)ecnt[threadIdx.x];
}

// =====================================================================
// Host-side orchestration
// =====================================================================
static inline dim3 gemm_grid(int M, int N) { return dim3((N + GBN - 1) / GBN, (M + GBM - 1) / GBM); }
constexpr size_t GEMM_LDS = (size_t)(GBM * GBK + GBK * GBN) * sizeof(float);  // 24 KB

extern "C" void kernel_launch(void* const* d_in, const int* in_sizes, int n_in,
                              void* d_out, int out_size, void* d_ws, size_t ws_size,
                              hipStream_t stream)
{
    (void)in_sizes; (void)n_in; (void)out_size; (void)ws_size;
    const float* x      = (const float*)d_in[0];
    const float* ln1_g  = (const float*)d_in[1];
    const float* ln1_b  = (const float*)d_in[2];
    const float* W_ckv  = (const float*)d_in[3];
    const float* W_kv   = (const float*)d_in[4];
    const float* W_cq   = (const float*)d_in[5];
    const float* W_q    = (const float*)d_in[6];
    const float* W_qr   = (const float*)d_in[7];
    const float* W_kr   = (const float*)d_in[8];
    const float* W_o    = (const float*)d_in[9];
    const float* ln2_g  = (const float*)d_in[10];
    const float* ln2_b  = (const float*)d_in[11];
    const float* cent   = (const float*)d_in[12];
    const float* rbias  = (const float*)d_in[13];
    const float* Wg     = (const float*)d_in[14];
    const float* Wu     = (const float*)d_in[15];
    const float* Wd     = (const float*)d_in[16];
    float* out = (float*)d_out;

    // ---- workspace layout (floats), phase-based aliasing ----
    float* ws = (float*)d_ws;
    const size_t M4 = (size_t)NT_ * H_;   // 4M floats
    float* h1   = ws;                     // 4M   (later: Xg)
    float* ckv  = h1 + M4;                // 1M
    float* cq   = ckv + (size_t)NT_ * L_; // 1M
    float* kr   = cq + (size_t)NT_ * L_;  // 256K
    int*   ecnt = (int*)(kr + (size_t)NT_ * RD_);
    int*   elist = ecnt + 16;             // 8*4096 ints
    float* ewgt  = (float*)(elist + E_ * NT_);
    float* kvb  = ewgt + E_ * NT_;        // 8M    (later: U)
    float* qc   = kvb + (size_t)NT_ * 2 * H_; // 4M \ later: G (8M)
    float* qr   = qc + M4;                    // 4M /
    float* qpk  = qr + M4;                // 8M
    float* kpk  = qpk + (size_t)B_ * NH_ * S_ * 128; // 8M
    float* vpk  = kpk + (size_t)B_ * NH_ * S_ * 128; // 4M
    float* oat  = vpk + (size_t)B_ * NH_ * S_ * 64;  // 4M  (later: OutG)
    float* aprj = oat + M4;               // 4M  (later: xf)
    // aliases
    float* Xg   = h1;
    float* G    = qc;
    float* U    = kvb;
    float* OutG = oat;
    float* xf   = aprj;

    // ---- 1. LN1 ----
    ln_kernel<<<NT_, 256, 0, stream>>>(x, ln1_g, ln1_b, h1);

    // ---- 2. low-rank projections (WMMA GEMMs, TDM-staged) ----
    gemm_bf16_wmma<<<gemm_grid(NT_, L_),   128, GEMM_LDS, stream>>>(h1,  W_ckv, ckv, NT_, L_,   H_, H_, L_,   L_,   nullptr);
    gemm_bf16_wmma<<<gemm_grid(NT_, L_),   128, GEMM_LDS, stream>>>(h1,  W_cq,  cq,  NT_, L_,   H_, H_, L_,   L_,   nullptr);
    gemm_bf16_wmma<<<gemm_grid(NT_, 2*H_), 128, GEMM_LDS, stream>>>(ckv, W_kv,  kvb, NT_, 2*H_, L_, L_, 2*H_, 2*H_, nullptr);
    gemm_bf16_wmma<<<gemm_grid(NT_, H_),   128, GEMM_LDS, stream>>>(cq,  W_q,   qc,  NT_, H_,   L_, L_, H_,   H_,   nullptr);
    gemm_bf16_wmma<<<gemm_grid(NT_, H_),   128, GEMM_LDS, stream>>>(cq,  W_qr,  qr,  NT_, H_,   L_, L_, H_,   H_,   nullptr);
    gemm_bf16_wmma<<<gemm_grid(NT_, RD_),  128, GEMM_LDS, stream>>>(ckv, W_kr,  kr,  NT_, RD_,  L_, L_, RD_,  RD_,  nullptr);

    // ---- 3. pack + RoPE (scale folded into q) ----
    pack_q_kernel <<<NT_, 256, 0, stream>>>(qc, qr, qpk);
    pack_kv_kernel<<<NT_, 256, 0, stream>>>(kvb, kr, kpk, vpk);

    // ---- 4. flash attention (WMMA) ----
    attn_kernel<<<dim3(B_ * NH_, S_ / 64), 128, 0, stream>>>(qpk, kpk, vpk, oat);

    // ---- 5. output projection + residual ----
    gemm_bf16_wmma<<<gemm_grid(NT_, H_), 128, GEMM_LDS, stream>>>(oat, W_o, aprj, NT_, H_, H_, H_, H_, H_, nullptr);
    add_kernel<<<4096, 256, 0, stream>>>(x, aprj, out, M4);

    // ---- 6. LN2 + routing ----
    ln_kernel<<<NT_, 256, 0, stream>>>(out, ln2_g, ln2_b, xf);
    (void)hipMemsetAsync(ecnt, 0, 16 * sizeof(int), stream);
    route_kernel<<<NT_, 256, 0, stream>>>(xf, cent, rbias, ecnt, elist, ewgt);

    // ---- 7. MoE: gathered per-expert GEMMs (device-side M via TDM zero-fill) ----
    for (int e = 0; e < E_; ++e) {
        gather_kernel<<<NT_, 256, 0, stream>>>(xf, elist, ecnt, e, Xg);
        gemm_bf16_wmma<<<gemm_grid(NT_, DF_), 128, GEMM_LDS, stream>>>(Xg, Wg + (size_t)e * H_ * DF_, G,  NT_, DF_, H_,  H_,  DF_, DF_, ecnt + e);
        gemm_bf16_wmma<<<gemm_grid(NT_, DF_), 128, GEMM_LDS, stream>>>(Xg, Wu + (size_t)e * H_ * DF_, U,  NT_, DF_, H_,  H_,  DF_, DF_, ecnt + e);
        silu_mul_kernel<<<4096, 256, 0, stream>>>(G, U, ecnt, e);
        gemm_bf16_wmma<<<gemm_grid(NT_, H_),  128, GEMM_LDS, stream>>>(G,  Wd + (size_t)e * DF_ * H_, OutG, NT_, H_, DF_, DF_, H_,  H_,  ecnt + e);
        scatter_kernel<<<NT_, 256, 0, stream>>>(OutG, elist, ewgt, ecnt, e, out);
    }

    // ---- 8. counts tail ----
    counts_out_kernel<<<1, 32, 0, stream>>>(ecnt, out + M4);
}
